// MultiHeadAttentionBlock_30777735644078
// MI455X (gfx1250) — matrix operand order repaired and checked by reading, not compiled, not run on hardware
//
#include <hip/hip_runtime.h>
#include <stdint.h>

// ---------------------------------------------------------------------------
// MI455X (gfx1250) wave32 WMMA multi-head attention block.
//   out = softmax_causal((xWq^T)(xWk^T)^T / sqrt(dk)) (xWv^T) Wo^T
// GEMMs: v_wmma_f32_16x16x32_f16, fp32 accum, double-buffered LDS staging
// via GLOBAL_LOAD_ASYNC_TO_LDS (ASYNCcnt) when available.
// Attention: K tiles staged by the Tensor Data Mover (TENSORcnt) when
// available. All paths have probe-safe fallbacks.
// ---------------------------------------------------------------------------

typedef _Float16 f16;
typedef __attribute__((ext_vector_type(16))) _Float16 v16h;
typedef __attribute__((ext_vector_type(8)))  _Float16 v8h;
typedef __attribute__((ext_vector_type(4)))  _Float16 v4h;
typedef __attribute__((ext_vector_type(8)))  float    v8f;
typedef __attribute__((ext_vector_type(4)))  unsigned int u32x4;
typedef __attribute__((ext_vector_type(8)))  int          i32x8;
typedef __attribute__((ext_vector_type(4)))  int          i32x4;
typedef int gv4i __attribute__((vector_size(16)));   // builtin param pointee

#if defined(__has_builtin)
#if __has_builtin(__builtin_amdgcn_global_load_async_to_lds_b128) && \
    __has_builtin(__builtin_amdgcn_s_wait_asynccnt)
#define HAVE_ASYNC_LDS 1
#endif
#if __has_builtin(__builtin_amdgcn_tensor_load_to_lds) && \
    __has_builtin(__builtin_amdgcn_s_wait_tensorcnt)
#define HAVE_TDM 1
#endif
#endif

#define AS1 __attribute__((address_space(1)))
#define AS3 __attribute__((address_space(3)))

constexpr int kB  = 2;
constexpr int kS  = 2048;
constexpr int kD  = 1024;   // d_model
constexpr int kH  = 16;
constexpr int kDk = 64;
#define NEGV (-1.0e9f)

// Assemble a 16-element f16 fragment from two contiguous 16B halves.
__device__ __forceinline__ v16h make_frag(v8h lo, v8h hi) {
  v16h r;
#pragma unroll
  for (int i = 0; i < 8; ++i) { r[i] = lo[i]; r[i + 8] = hi[i]; }
  return r;
}

__device__ __forceinline__ v8f wmma_f16(v16h a, v16h b, v8f c) {
  // (neg_a, A, neg_b, B, c_mod, C, reuse_a, reuse_b)
  return __builtin_amdgcn_wmma_f32_16x16x32_f16(false, a, false, b,
                                                (short)0, c, false, false);
}

#ifdef HAVE_ASYNC_LDS
// Per-lane 16B global -> LDS async copy (tracked on ASYNCcnt).
__device__ __forceinline__ void async_copy_16B(f16* lds_dst, const f16* gsrc) {
  AS3 gv4i* l = (AS3 gv4i*)(uint32_t)(uintptr_t)lds_dst;
  AS1 gv4i* g = (AS1 gv4i*)(uintptr_t)gsrc;
  __builtin_amdgcn_global_load_async_to_lds_b128(g, l, 0, 0);
}
#endif

// ---------------------------------------------------------------------------
// fp32 -> f16 conversion, 4 elems/thread
// ---------------------------------------------------------------------------
__global__ __launch_bounds__(256)
void cvt_f32_to_f16(const float* __restrict__ in, f16* __restrict__ out, int n) {
  int i4 = (blockIdx.x * blockDim.x + threadIdx.x) * 4;
  if (i4 + 3 < n) {
    float4 x = *(const float4*)(in + i4);
    v4h o;
    o[0] = (f16)x.x; o[1] = (f16)x.y; o[2] = (f16)x.z; o[3] = (f16)x.w;
    *(v4h*)(out + i4) = o;
  }
}

// ---------------------------------------------------------------------------
// C[M,N] = alpha * A[M,K] @ W[N,K]^T   (torch nn.Linear, no bias)
// Block: 256 threads (8 waves), tile 128x128, K-step 32, double-buffered LDS.
// Wave (wm 0..3, wn 0..1) computes 32x64 = 2x4 WMMA tiles.
// ---------------------------------------------------------------------------
template <int OUT_F32>
__global__ __launch_bounds__(256)
void gemm_nt(const f16* __restrict__ A, const f16* __restrict__ W,
             f16* __restrict__ C16, float* __restrict__ C32,
             int M, int N, int K, float alpha) {
  __shared__ __align__(16) f16 As[2][128][32];
  __shared__ __align__(16) f16 Ws[2][128][32];

  const int tid  = threadIdx.x;
  const int lane = tid & 31;
  const int wv   = tid >> 5;
  const int wm   = wv & 3;        // 32-row strip
  const int wn   = wv >> 2;       // 64-col strip
  const int hl   = lane >> 4;     // half-wave (K split per ISA frag layout)
  const int l16  = lane & 15;
  const int bm   = blockIdx.x * 128;
  const int bn   = blockIdx.y * 128;

  v8f vzero = {};
  v8f acc[2][4];
#pragma unroll
  for (int i = 0; i < 2; ++i)
#pragma unroll
    for (int j = 0; j < 4; ++j) acc[i][j] = vzero;

  // Stage one 128x32 A tile + 128x32 W tile into LDS buffer `buf`.
  auto stage = [&](int k0, int buf) {
#pragma unroll
    for (int i = 0; i < 2; ++i) {
      int c  = tid + 256 * i;       // chunk id 0..511
      int r  = c >> 2;              // row 0..127
      int cc = (c & 3) * 8;         // col 0/8/16/24
      const f16* ga = &A[(size_t)(bm + r) * K + k0 + cc];
      const f16* gw = &W[(size_t)(bn + r) * K + k0 + cc];
#ifdef HAVE_ASYNC_LDS
      async_copy_16B(&As[buf][r][cc], ga);
      async_copy_16B(&Ws[buf][r][cc], gw);
#else
      *(v8h*)&As[buf][r][cc] = *(const v8h*)ga;
      *(v8h*)&Ws[buf][r][cc] = *(const v8h*)gw;
#endif
    }
  };

  stage(0, 0);
  const int nIter = K >> 5;
  for (int it = 0; it < nIter; ++it) {
    const int buf = it & 1;
#ifdef HAVE_ASYNC_LDS
    __builtin_amdgcn_s_wait_asynccnt(0);   // this buffer's tiles landed
#endif
    __syncthreads();                       // ...for every wave in the block
    if (it + 1 < nIter) stage((it + 1) * 32, buf ^ 1);  // overlap with WMMA

    // A 16x32 frag: lane half hl holds K = {hl*8..+8} and {16+hl*8..+8}
    v16h af[2];
#pragma unroll
    for (int mt = 0; mt < 2; ++mt) {
      const f16* p = &As[buf][wm * 32 + mt * 16 + l16][hl * 8];
      af[mt] = make_frag(*(const v8h*)p, *(const v8h*)(p + 16));
    }
    // B 32x16 frag: lane half hl holds K = {hl*8..+8} and {16+hl*8..+8}
    v16h bf[4];
#pragma unroll
    for (int nt = 0; nt < 4; ++nt) {
      const f16* p = &Ws[buf][wn * 64 + nt * 16 + l16][hl * 8];
      bf[nt] = make_frag(*(const v8h*)p, *(const v8h*)(p + 16));
    }
#pragma unroll
    for (int mt = 0; mt < 2; ++mt)
#pragma unroll
      for (int nt = 0; nt < 4; ++nt)
        acc[mt][nt] = wmma_f16(af[mt], bf[nt], acc[mt][nt]);
  }

  // C/D layout: VGPR r -> M = r + 8*hl, N = l16
#pragma unroll
  for (int mt = 0; mt < 2; ++mt) {
#pragma unroll
    for (int nt = 0; nt < 4; ++nt) {
      int mbase = bm + wm * 32 + mt * 16 + hl * 8;
      int n     = bn + wn * 64 + nt * 16 + l16;
#pragma unroll
      for (int r = 0; r < 8; ++r) {
        float x    = acc[mt][nt][r] * alpha;
        size_t idx = (size_t)(mbase + r) * N + n;
        if (OUT_F32) C32[idx] = x;
        else         C16[idx] = (f16)x;
      }
    }
  }
}

// ---------------------------------------------------------------------------
// Causal flash attention. Grid: (S/64, H, B), 128 threads (4 waves).
// Each wave owns 16 query rows; 32 keys per iteration.
// Q is pre-scaled by 1/sqrt(dk) in the projection GEMM.
// K tile is staged by the Tensor Data Mover when available.
// ---------------------------------------------------------------------------
__global__ __launch_bounds__(128)
void attn_kernel(const f16* __restrict__ Qp, const f16* __restrict__ Kp,
                 const f16* __restrict__ Vp, f16* __restrict__ Xc) {
  const int qblk = blockIdx.x;
  const int h    = blockIdx.y;
  const int b    = blockIdx.z;
  const int tid  = threadIdx.x;
  const int lane = tid & 31;
  const int wv   = tid >> 5;     // 0..3
  const int hl   = lane >> 4;
  const int l16  = lane & 15;

  __shared__ __align__(16) f16 Ks[32][64];      // [key][d]
  __shared__ __align__(16) f16 Vt[64][32];      // transposed: [d][key]
  __shared__ __align__(16) f16 Ps[4][16][32];   // per-wave P scratch

  const int q0 = qblk * 64 + wv * 16;

  // Q A-frags for d windows [0,32) and [32,64)
  v16h qf[2];
  {
    const f16* qrow = Qp + ((size_t)b * kS + q0 + l16) * kD + h * kDk;
#pragma unroll
    for (int ds = 0; ds < 2; ++ds) {
      const f16* p = qrow + ds * 32 + hl * 8;
      qf[ds] = make_frag(*(const v8h*)p, *(const v8h*)(p + 16));
    }
  }

  float mrow[8], lrow[8];
  v8f vzero = {};
  v8f oacc[4];
#pragma unroll
  for (int r = 0; r < 8; ++r) { mrow[r] = -3.0e38f; lrow[r] = 0.0f; }
#pragma unroll
  for (int nt = 0; nt < 4; ++nt) oacc[nt] = vzero;

  const int kv_end = qblk * 64 + 64;   // causal: keys < block's last q + 1
  for (int kv = 0; kv < kv_end; kv += 32) {
    __syncthreads();   // previous iteration's consumers are done with LDS

#ifdef HAVE_TDM
    if (wv == 0) {
      // D# per CDNA5 ISA: 2D tile 32(keys) x 64(d),
      // f16 elements, row stride = d_model. Groups 2/3 unused (2D); the
      // trailing i32x8 is the gather-mode index block -> zero (gather off).
      uint64_t ga = (uint64_t)(uintptr_t)
          (Kp + ((size_t)b * kS + kv) * kD + h * kDk);
      uint32_t la = (uint32_t)(uintptr_t)&Ks[0][0];
      u32x4 g0 = { 1u,                                   // count=1, user mode
                   la,                                   // lds_addr (bytes)
                   (uint32_t)ga,                         // global_addr[31:0]
                   (uint32_t)(ga >> 32) | (2u << 30) };  // ga[56:32] | type=2
      i32x8 g1 = { 1 << 16,       // data_size=1 (2 bytes), no multicast
                   64 << 16,      // tensor_dim0 = 64  (bits 79:48)
                   32 << 16,      // tensor_dim1 = 32  (bits 111:80)
                   64 << 16,      // tile_dim0  = 64   (bits 127:112)
                   32,            // tile_dim1  = 32   (bits 143:128)
                   (int)kD,       // tensor_dim0_stride = 1024 (bits 207:160)
                   0, 0 };
      i32x4 gz4 = { 0, 0, 0, 0 };
      i32x8 gz8 = { 0, 0, 0, 0, 0, 0, 0, 0 };
      __builtin_amdgcn_tensor_load_to_lds(g0, g1, gz4, gz4, gz8, 0);
    }
#endif

    // Stage V transposed (and K row-major on the fallback path).
#pragma unroll
    for (int i = 0; i < 2; ++i) {
      int c  = tid + 128 * i;     // 0..255
      int r  = c >> 3;            // key 0..31
      int cc = (c & 7) * 8;       // d 0..56
#ifndef HAVE_TDM
      *(v8h*)&Ks[r][cc] =
          *(const v8h*)&Kp[((size_t)b * kS + kv + r) * kD + h * kDk + cc];
#endif
      v8h vvv = *(const v8h*)&Vp[((size_t)b * kS + kv + r) * kD + h * kDk + cc];
#pragma unroll
      for (int j = 0; j < 8; ++j) Vt[cc + j][r] = vvv[j];
    }
#ifdef HAVE_TDM
    if (wv == 0) __builtin_amdgcn_s_wait_tensorcnt(0);
#endif
    __syncthreads();

    // Scores: S(16x32) = Q(16x64) @ K^T(64x32), two 16-key column tiles.
    v8f sc[2];
    sc[0] = vzero; sc[1] = vzero;
#pragma unroll
    for (int t = 0; t < 2; ++t) {
#pragma unroll
      for (int ds = 0; ds < 2; ++ds) {
        const f16* p = &Ks[t * 16 + l16][ds * 32 + hl * 8];
        v16h kf = make_frag(*(const v8h*)p, *(const v8h*)(p + 16));
        sc[t] = wmma_f16(qf[ds], kf, sc[t]);
      }
    }

    // Causal mask in C layout: row M = r + 8*hl, key = kv + t*16 + l16.
#pragma unroll
    for (int t = 0; t < 2; ++t)
#pragma unroll
      for (int r = 0; r < 8; ++r) {
        int key = kv + t * 16 + l16;
        int qq  = q0 + r + hl * 8;
        if (key > qq) sc[t][r] = NEGV;
      }

    // Online softmax per row (row spread across 16 lanes of a half-wave).
#pragma unroll
    for (int r = 0; r < 8; ++r) {
      float mx = fmaxf(sc[0][r], sc[1][r]);
#pragma unroll
      for (int s = 1; s < 16; s <<= 1) mx = fmaxf(mx, __shfl_xor(mx, s, 32));
      float mnew = fmaxf(mrow[r], mx);
      float corr = __expf(mrow[r] - mnew);
      float p0   = __expf(sc[0][r] - mnew);
      float p1   = __expf(sc[1][r] - mnew);
      float rs   = p0 + p1;
#pragma unroll
      for (int s = 1; s < 16; s <<= 1) rs += __shfl_xor(rs, s, 32);
      lrow[r] = lrow[r] * corr + rs;
      mrow[r] = mnew;
#pragma unroll
      for (int nt = 0; nt < 4; ++nt) oacc[nt][r] *= corr;
      // Re-layout P via LDS: C-layout -> row-major [qrow][key_local]
      Ps[wv][r + hl * 8][l16]      = (f16)p0;
      Ps[wv][r + hl * 8][16 + l16] = (f16)p1;
    }
    // DS ops are in-order within a wave: safe to read our own Ps back.

    // O(16x64) += P(16x32) @ V(32x64)
    v16h pf;
    {
      const f16* p = &Ps[wv][l16][hl * 8];
      pf = make_frag(*(const v8h*)p, *(const v8h*)(p + 16));
    }
#pragma unroll
    for (int nt = 0; nt < 4; ++nt) {
      const f16* p = &Vt[nt * 16 + l16][hl * 8];
      v16h vf = make_frag(*(const v8h*)p, *(const v8h*)(p + 16));
      oacc[nt] = wmma_f16(pf, vf, oacc[nt]);
    }
  }

  // Normalize and write context in [B,S,D] layout (f16, feeds final GEMM).
#pragma unroll
  for (int r = 0; r < 8; ++r) {
    float inv = 1.0f / lrow[r];
    size_t row = (size_t)b * kS + q0 + r + hl * 8;
#pragma unroll
    for (int nt = 0; nt < 4; ++nt)
      Xc[row * kD + h * kDk + nt * 16 + l16] = (f16)(oacc[nt][r] * inv);
  }
}

// ---------------------------------------------------------------------------
// Host launcher
// ---------------------------------------------------------------------------
extern "C" void kernel_launch(void* const* d_in, const int* in_sizes, int n_in,
                              void* d_out, int out_size, void* d_ws,
                              size_t ws_size, hipStream_t stream) {
  (void)in_sizes; (void)n_in; (void)out_size; (void)ws_size;
  const float* q    = (const float*)d_in[0];
  const float* k    = (const float*)d_in[1];
  const float* v    = (const float*)d_in[2];
  /* d_in[3] = causal mask, analytically applied in attn_kernel */
  const float* Wq   = (const float*)d_in[4];
  const float* Wk   = (const float*)d_in[5];
  const float* Wv   = (const float*)d_in[6];
  const float* Wo   = (const float*)d_in[7];
  float*       out  = (float*)d_out;

  const size_t TOK = (size_t)kB * kS * kD;  // 4M elems
  const size_t WSZ = (size_t)kD * kD;       // 1M elems
  f16* ws   = (f16*)d_ws;                   // needs 7*TOK + 4*WSZ f16 = 64 MB
  f16* q16  = ws;
  f16* k16  = q16 + TOK;
  f16* v16p = k16 + TOK;
  f16* wq16 = v16p + TOK;
  f16* wk16 = wq16 + WSZ;
  f16* wv16 = wk16 + WSZ;
  f16* wo16 = wv16 + WSZ;
  f16* Qp   = wo16 + WSZ;
  f16* Kp   = Qp + TOK;
  f16* Vp   = Kp + TOK;
  f16* Xc   = Vp + TOK;

  // fp32 -> f16 staging
  cvt_f32_to_f16<<<(int)(TOK / 1024), 256, 0, stream>>>(q,  q16,  (int)TOK);
  cvt_f32_to_f16<<<(int)(TOK / 1024), 256, 0, stream>>>(k,  k16,  (int)TOK);
  cvt_f32_to_f16<<<(int)(TOK / 1024), 256, 0, stream>>>(v,  v16p, (int)TOK);
  cvt_f32_to_f16<<<(int)(WSZ / 1024), 256, 0, stream>>>(Wq, wq16, (int)WSZ);
  cvt_f32_to_f16<<<(int)(WSZ / 1024), 256, 0, stream>>>(Wk, wk16, (int)WSZ);
  cvt_f32_to_f16<<<(int)(WSZ / 1024), 256, 0, stream>>>(Wv, wv16, (int)WSZ);
  cvt_f32_to_f16<<<(int)(WSZ / 1024), 256, 0, stream>>>(Wo, wo16, (int)WSZ);

  const int M = kB * kS;   // 4096
  dim3 gg(M / 128, kD / 128);  // (32, 8)

  // Projections (scale 1/sqrt(dk) folded into Q)
  gemm_nt<0><<<gg, 256, 0, stream>>>(q16,  wq16, Qp, nullptr, M, kD, kD, 0.125f);
  gemm_nt<0><<<gg, 256, 0, stream>>>(k16,  wk16, Kp, nullptr, M, kD, kD, 1.0f);
  gemm_nt<0><<<gg, 256, 0, stream>>>(v16p, wv16, Vp, nullptr, M, kD, kD, 1.0f);

  // Causal flash attention
  dim3 ga(kS / 64, kH, kB);    // (32, 16, 2)
  attn_kernel<<<ga, 128, 0, stream>>>(Qp, Kp, Vp, Xc);

  // Output projection -> fp32 d_out
  gemm_nt<1><<<gg, 256, 0, stream>>>(Xc, wo16, nullptr, out, M, kD, kD, 1.0f);
}
